// tattn_block_85676007621019
// MI455X (gfx1250) — matrix operand order repaired and checked by reading, not compiled, not run on hardware
//
#include <hip/hip_runtime.h>
#include <math.h>

typedef __bf16 bf16;
typedef __attribute__((ext_vector_type(16))) __bf16 v16bf;
typedef __attribute__((ext_vector_type(8)))  __bf16 v8bf;
typedef __attribute__((ext_vector_type(8)))  float  v8f;

#define NTOK 96000   // B*V*T
#define HDIM 256
#define TLEN 120
#define QKVD 768

__device__ __forceinline__ v16bf cat8(v8bf lo, v8bf hi) {
  return __builtin_shufflevector(lo, hi, 0,1,2,3,4,5,6,7,8,9,10,11,12,13,14,15);
}

// gfx1250 async copy: global -> LDS, 16B per lane, tracked by ASYNCcnt.
// VDST operand carries the LDS byte address (= low 32 bits of the generic
// shared pointer; ISA 10.2: LDS aperture addr[31:0] is the LDS address).
__device__ __forceinline__ void async_load_b128(const bf16* gsrc, bf16* lds) {
  unsigned ldsaddr = (unsigned)(size_t)lds;
  asm volatile("global_load_async_to_lds_b128 %0, %1, off"
               :: "v"(ldsaddr), "v"(gsrc) : "memory");
}
__device__ __forceinline__ void wait_asynccnt0() {
  asm volatile("s_wait_asynccnt 0" ::: "memory");
}

// ---------------------------------------------------------------------------
// WMMA GEMM: C[M,N] = A[M,K](bf16) * Bt[N,K]^T(bf16), f32 accumulate.
// Block = 256 threads = 8 waves stacked along M (block tile 128x64).
// B-tile (64 x KC) staged in LDS via async DMA, double buffered; A fragments
// stream from global. k-loop software-pipelined: next fragments load while
// the current 4 WMMAs issue.
// MODE 0: outF = v*e0[col] + e1[col]                  (pre_proj BN)
// MODE 1: outB = bf16(v + e0[col])                    (qkv bias)
// MODE 2: outF += v + e0[col]                         (residual adds)
// MODE 3: outB = bf16(gelu(v + e0[col]))              (fc1)
// MODE 4: scatter outF[b,col,t,v] = v + e0[col]       (folded tail)
// ---------------------------------------------------------------------------
template<int MODE, int KC>
__global__ __launch_bounds__(256)
void gemm_bf16_wmma(const bf16* __restrict__ A, int lda,
                    const bf16* __restrict__ Bt, int ldb,
                    int M, int N, int K,
                    const float* __restrict__ e0, const float* __restrict__ e1,
                    float* __restrict__ outF, bf16* __restrict__ outB, int ldo)
{
  constexpr int SLD = KC + 8;                 // +16B pad per row: bank spread
  __shared__ bf16 sB[2][64 * (128 + 8)];

  const int lane = threadIdx.x & 31;
  const int wave = threadIdx.x >> 5;
  const int tiles_n = N >> 6;
  const int bm = blockIdx.x / tiles_n;
  const int bn = blockIdx.x - bm * tiles_n;
  const int m0 = (bm * 8 + wave) << 4;
  const int n0 = bn << 6;

  const int l15 = lane & 15;
  const int hiL = lane >> 4;                  // 0: lanes 0-15, 1: lanes 16-31
  const int koffA = hiL ? 8 : 0;              // A frag: K[0..7]/[16..23] vs K[8..15]/[24..31]
  const int koffB = hiL ? 8 : 0;              // B frag: K[0..7]/[16..23] vs K[8..15]/[24..31]

  const bf16* aRow = A + (size_t)(m0 + l15) * lda + koffA;

  auto stage = [&](int chunk, int buf) {      // cooperative DMA of Bt tile
    const int kc = chunk * KC;
    constexpr unsigned per_row = KC / 8;      // 16B units per row
    constexpr int iters = (8 * KC) / 256;     // 4 (KC=128) or 2 (KC=64)
    #pragma unroll
    for (int r = 0; r < iters; ++r) {
      unsigned i   = (unsigned)threadIdx.x + (unsigned)r * 256u;
      unsigned row = i / per_row;
      unsigned ce  = (i % per_row) * 8u;      // bf16 element offset in row
      async_load_b128(Bt + (size_t)(n0 + row) * ldb + kc + ce,
                      &sB[buf][row * SLD + ce]);
    }
  };

  auto loadB = [&](const bf16* sb, int k, int nn) -> v16bf {
    const bf16* p = sb + (l15 + nn * 16) * SLD + k + koffB;
    return cat8(*(const v8bf*)p, *(const v8bf*)(p + 16));
  };
  auto loadA = [&](int kk) -> v16bf {
    return cat8(*(const v8bf*)(aRow + kk), *(const v8bf*)(aRow + kk + 16));
  };

  v8f c0 = {}, c1 = {}, c2 = {}, c3 = {};
  const int nch = K / KC;
  stage(0, 0);
  for (int c = 0; c < nch; ++c) {
    wait_asynccnt0();                         // own DMA done
    __syncthreads();                          // tile published; prev reads retired
    if (c + 1 < nch) stage(c + 1, (c + 1) & 1);
    const bf16* sb = &sB[c & 1][0];
    const int kc = c * KC;

    v16bf af = loadA(kc);
    v16bf b0 = loadB(sb, 0, 0), b1 = loadB(sb, 0, 1),
          b2 = loadB(sb, 0, 2), b3 = loadB(sb, 0, 3);
    #pragma unroll
    for (int k = 0; k < KC; k += 32) {
      v16bf afn = af, bn0 = b0, bn1 = b1, bn2 = b2, bn3 = b3;
      if (k + 32 < KC) {                      // compile-time per unrolled iter
        afn = loadA(kc + k + 32);
        bn0 = loadB(sb, k + 32, 0);
        bn1 = loadB(sb, k + 32, 1);
        bn2 = loadB(sb, k + 32, 2);
        bn3 = loadB(sb, k + 32, 3);
      }
      c0 = __builtin_amdgcn_wmma_f32_16x16x32_bf16(false, af, false, b0, (short)0, c0, false, false);
      c1 = __builtin_amdgcn_wmma_f32_16x16x32_bf16(false, af, false, b1, (short)0, c1, false, false);
      c2 = __builtin_amdgcn_wmma_f32_16x16x32_bf16(false, af, false, b2, (short)0, c2, false, false);
      c3 = __builtin_amdgcn_wmma_f32_16x16x32_bf16(false, af, false, b3, (short)0, c3, false, false);
      af = afn; b0 = bn0; b1 = bn1; b2 = bn2; b3 = bn3;
    }
  }

  // C/D layout: VGPR r, lanes 0-15 -> (M=r, N=lane); lanes 16-31 -> (M=8+r, N=lane-16)
  const int rbase = m0 + (hiL ? 8 : 0);
  v8f accs[4] = {c0, c1, c2, c3};
  #pragma unroll
  for (int nn = 0; nn < 4; ++nn) {
    const int col = n0 + nn * 16 + l15;
    #pragma unroll
    for (int r = 0; r < 8; ++r) {
      const int row = rbase + r;
      float v = accs[nn][r];
      if constexpr (MODE == 0) {
        outF[(size_t)row * ldo + col] = v * e0[col] + e1[col];
      } else if constexpr (MODE == 1) {
        outB[(size_t)row * ldo + col] = (bf16)(v + e0[col]);
      } else if constexpr (MODE == 2) {
        outF[(size_t)row * ldo + col] += v + e0[col];
      } else if constexpr (MODE == 3) {
        float g = v + e0[col];
        outB[(size_t)row * ldo + col] =
            (bf16)(0.5f * g * (1.0f + erff(g * 0.70710678118654752f)));
      } else { // MODE 4: token row -> (b, t, vv)
        int t  = row % TLEN;
        int vv = (row / TLEN) % 25;
        int b  = row / 3000;
        outF[(size_t)((b * 64 + col) * TLEN + t) * 25 + vv] = v + e0[col];
      }
    }
  }
}

// --------------------------- LayerNorm (wave per token) --------------------
__global__ __launch_bounds__(256)
void ln_kernel(const float* __restrict__ in, const float* __restrict__ g,
               const float* __restrict__ b, bf16* __restrict__ out)
{
  const int lane = threadIdx.x & 31;
  const int tok  = blockIdx.x * 8 + (threadIdx.x >> 5);
  const float* p = in + (size_t)tok * HDIM;
  float x[8], s = 0.f, s2 = 0.f;
  #pragma unroll
  for (int j = 0; j < 8; ++j) { x[j] = p[lane * 8 + j]; s += x[j]; s2 += x[j] * x[j]; }
  #pragma unroll
  for (int off = 16; off > 0; off >>= 1) {
    s  += __shfl_xor(s,  off, 32);
    s2 += __shfl_xor(s2, off, 32);
  }
  const float mean = s * (1.f / HDIM);
  const float inv  = rsqrtf(s2 * (1.f / HDIM) - mean * mean + 1e-6f);
  bf16* q = out + (size_t)tok * HDIM;
  #pragma unroll
  for (int j = 0; j < 8; ++j) {
    int c = lane * 8 + j;
    q[c] = (bf16)((x[j] - mean) * inv * g[c] + b[c]);
  }
}

// ---------------- windowed attention: thread per (n, window, head, query) --
__global__ __launch_bounds__(128)
void attn_kernel(const bf16* __restrict__ qkv, bf16* __restrict__ o)
{
  const int tid = blockIdx.x * 128 + threadIdx.x;
  if (tid >= 800 * 20 * 4 * 6) return;
  const int qi = tid % 6;
  const int hD = (tid / 6) % 4;
  const int w  = (tid / 24) % 20;
  const int n  = tid / 480;
  const int mbase = n * TLEN + w * 6;

  const bf16* qp = qkv + (size_t)(mbase + qi) * QKVD + hD * 64;
  float q[64];
  #pragma unroll
  for (int d = 0; d < 64; ++d) q[d] = (float)qp[d];

  float s[6], mx = -1e30f;
  for (int j = 0; j < 6; ++j) {
    const bf16* kp = qkv + (size_t)(mbase + j) * QKVD + 256 + hD * 64;
    float acc = 0.f;
    #pragma unroll
    for (int d = 0; d < 64; ++d) acc += q[d] * (float)kp[d];
    s[j] = acc * 0.125f;                       // HD^-0.5
    mx = fmaxf(mx, s[j]);
  }
  float den = 0.f;
  #pragma unroll
  for (int j = 0; j < 6; ++j) { s[j] = __expf(s[j] - mx); den += s[j]; }
  const float rden = 1.f / den;

  bf16* op = o + (size_t)(mbase + qi) * HDIM + hD * 64;
  for (int d = 0; d < 64; ++d) {
    float acc = 0.f;
    #pragma unroll
    for (int j = 0; j < 6; ++j)
      acc += s[j] * (float)qkv[(size_t)(mbase + j) * QKVD + 512 + hD * 64 + d];
    op[d] = (bf16)(acc * rden);
  }
}

// ----------- PEG: depthwise 3-tap conv along T + bias + residual -----------
__global__ __launch_bounds__(256)
void peg_kernel(const float* __restrict__ T0, const float* __restrict__ pw,
                const float* __restrict__ pb, float* __restrict__ T1)
{
  const int tid = blockIdx.x * 256 + threadIdx.x;     // 96000*256
  const int h = tid & 255;
  const int m = tid >> 8;
  const int t = m % TLEN;
  // peg_w HWIO (3,3,1,256); only kw==1 column survives (W==1 + pad 1)
  float c = T0[(size_t)m * 256 + h] * pw[1024 + h];                // kh=1
  if (t > 0)   c += T0[(size_t)(m - 1) * 256 + h] * pw[256 + h];   // kh=0
  if (t < 119) c += T0[(size_t)(m + 1) * 256 + h] * pw[1792 + h];  // kh=2
  T1[tid] = T0[tid] + c + pb[h];
}

// ------------------------------ small helpers ------------------------------
__global__ void reorder_x_kernel(const float* __restrict__ x, bf16* __restrict__ xt)
{
  const int tid = blockIdx.x * 256 + threadIdx.x;     // 96000*64
  const int c = tid & 63;
  const int m = tid >> 6;
  const int t = m % TLEN;
  const int vv = (m / TLEN) % 25;
  const int b = m / 3000;
  xt[tid] = (bf16)x[((size_t)(b * 64 + c) * TLEN + t) * 25 + vv];
}

__global__ void cast_f32_bf16_kernel(const float* __restrict__ in, bf16* __restrict__ out)
{
  const int tid = blockIdx.x * 256 + threadIdx.x;
  out[tid] = (bf16)in[tid];
}

__global__ void transpose_cast_kernel(const float* __restrict__ in, bf16* __restrict__ out,
                                      int K, int N)   // in: K x N (in-major) -> out: N x K
{
  const int tid = blockIdx.x * 256 + threadIdx.x;
  const int i = tid % K;
  const int o = tid / K;
  out[tid] = (bf16)in[(size_t)i * N + o];
}

// Fold BN/dw affines; compute folded tail bias.
__global__ void prep_kernel(const float* pre_g, const float* pre_b, const float* pre_m, const float* pre_v,
                            const float* post_g, const float* post_b, const float* post_m, const float* post_v,
                            const float* dw_w, const float* dw_b,
                            const float* pw_w, const float* pw_b,
                            const float* bn_g, const float* bn_b, const float* bn_m, const float* bn_v,
                            float* preS, float* preA, float* aV, float* s2V, float* cst)
{
  const int t = threadIdx.x;
  __shared__ float bV[64], t2[64];
  if (t < 256) {
    float s = pre_g[t] * rsqrtf(pre_v[t] + 1e-5f);
    preS[t] = s; preA[t] = pre_b[t] - pre_m[t] * s;
  }
  if (t < 64) {
    float s1 = post_g[t] * rsqrtf(post_v[t] + 1e-5f);
    aV[t] = dw_w[t] * s1;
    bV[t] = dw_w[t] * (post_b[t] - post_m[t] * s1) + dw_b[t];
    float s2 = bn_g[t] * rsqrtf(bn_v[t] + 1e-5f);
    s2V[t] = s2; t2[t] = bn_b[t] - bn_m[t] * s2;
  }
  __syncthreads();
  if (t < 64) {
    float acc = 0.f;
    for (int c = 0; c < 64; ++c) acc += pw_w[t * 64 + c] * bV[c];
    cst[t] = s2V[t] * (acc + pw_b[t]) + t2[t];
  }
}

// W_eff[o,h] = s2[o] * sum_c pw[o,c] * a[c] * post_w[c,h]   (64 x 256, bf16)
__global__ void weff_kernel(const float* __restrict__ post_w, const float* __restrict__ pw_w,
                            const float* __restrict__ aV, const float* __restrict__ s2V,
                            bf16* __restrict__ weff)
{
  const int h = threadIdx.x;      // 256
  const int o = blockIdx.x;       // 64
  float acc = 0.f;
  for (int c = 0; c < 64; ++c) acc += pw_w[o * 64 + c] * aV[c] * post_w[c * 256 + h];
  weff[o * 256 + h] = (bf16)(s2V[o] * acc);
}

// ---------------------------------------------------------------------------
extern "C" void kernel_launch(void* const* d_in, const int* in_sizes, int n_in,
                              void* d_out, int out_size, void* d_ws, size_t ws_size,
                              hipStream_t stream)
{
  const float* x      = (const float*)d_in[0];
  const float* pre_w  = (const float*)d_in[3];
  const float* ln1_g  = (const float*)d_in[8];
  const float* ln1_b  = (const float*)d_in[9];
  const float* qkv_w  = (const float*)d_in[10];
  const float* qkv_b  = (const float*)d_in[11];
  const float* apj_w  = (const float*)d_in[12];
  const float* apj_b  = (const float*)d_in[13];
  const float* peg_w  = (const float*)d_in[14];
  const float* peg_b  = (const float*)d_in[15];
  const float* ln2_g  = (const float*)d_in[16];
  const float* ln2_b  = (const float*)d_in[17];
  const float* fc1_w  = (const float*)d_in[18];
  const float* fc1_b  = (const float*)d_in[19];
  const float* fc2_w  = (const float*)d_in[20];
  const float* fc2_b  = (const float*)d_in[21];
  const float* post_w = (const float*)d_in[22];
  const float* pw_w   = (const float*)d_in[29];
  const float* pw_b   = (const float*)d_in[30];

  char* ws = (char*)d_ws;
  float* T0  = (float*)(ws);                          // 96000*256 f32
  float* T1  = (float*)(ws + 98304000);               // 96000*256 f32
  bf16*  BFX = (bf16*)(ws + 196608000);               // 96000*256 bf16
  bf16*  BFY = (bf16*)(ws + 245760000);               // 96000*1024 bf16
  bf16*  preW   = (bf16*)(ws + 442368000);            // 256*64
  bf16*  qkvT   = preW   + 16384;                     // 768*256
  bf16*  aprojT = qkvT   + 196608;                    // 256*256
  bf16*  fc1T   = aprojT + 65536;                     // 1024*256
  bf16*  fc2T   = fc1T   + 262144;                    // 256*1024
  bf16*  weff   = fc2T   + 262144;                    // 64*256
  float* preS = (float*)(ws + 444006400);
  float* preA = preS + 256;
  float* aV   = preA + 256;
  float* s2V  = aV + 64;
  float* cst  = s2V + 64;

  // ---- weight prep ----
  prep_kernel<<<1, 256, 0, stream>>>(
      (const float*)d_in[4], (const float*)d_in[5], (const float*)d_in[6], (const float*)d_in[7],
      (const float*)d_in[23], (const float*)d_in[24], (const float*)d_in[25], (const float*)d_in[26],
      (const float*)d_in[27], (const float*)d_in[28], pw_w, pw_b,
      (const float*)d_in[31], (const float*)d_in[32], (const float*)d_in[33], (const float*)d_in[34],
      preS, preA, aV, s2V, cst);
  weff_kernel<<<64, 256, 0, stream>>>(post_w, pw_w, aV, s2V, weff);
  cast_f32_bf16_kernel<<<64, 256, 0, stream>>>(pre_w, preW);            // 16384 (already out-major)
  transpose_cast_kernel<<<768, 256, 0, stream>>>(qkv_w, qkvT, 256, 768);
  transpose_cast_kernel<<<256, 256, 0, stream>>>(apj_w, aprojT, 256, 256);
  transpose_cast_kernel<<<1024, 256, 0, stream>>>(fc1_w, fc1T, 256, 1024);
  transpose_cast_kernel<<<1024, 256, 0, stream>>>(fc2_w, fc2T, 1024, 256);

  // ---- pipeline ----
  reorder_x_kernel<<<24000, 256, 0, stream>>>(x, BFY);                  // x -> tokens bf16 (96000x64)
  gemm_bf16_wmma<0, 64><<<750 * 4, 256, 0, stream>>>(BFY, 64, preW, 64,
      NTOK, 256, 64, preS, preA, T0, nullptr, 256);                     // pre_proj + BN -> T0
  ln_kernel<<<12000, 256, 0, stream>>>(T0, ln1_g, ln1_b, BFX);          // LN1
  gemm_bf16_wmma<1, 128><<<750 * 12, 256, 0, stream>>>(BFX, 256, qkvT, 256,
      NTOK, 768, 256, qkv_b, nullptr, nullptr, BFY, 768);               // qkv
  attn_kernel<<<3000, 128, 0, stream>>>(BFY, BFX);                      // windowed attention -> o
  gemm_bf16_wmma<2, 128><<<750 * 4, 256, 0, stream>>>(BFX, 256, aprojT, 256,
      NTOK, 256, 256, apj_b, nullptr, T0, nullptr, 256);                // attn proj + residual
  peg_kernel<<<96000, 256, 0, stream>>>(T0, peg_w, peg_b, T1);          // PEG
  ln_kernel<<<12000, 256, 0, stream>>>(T1, ln2_g, ln2_b, BFX);          // LN2
  gemm_bf16_wmma<3, 128><<<750 * 16, 256, 0, stream>>>(BFX, 256, fc1T, 256,
      NTOK, 1024, 256, fc1_b, nullptr, nullptr, BFY, 1024);             // fc1 + GELU
  gemm_bf16_wmma<2, 128><<<750 * 4, 256, 0, stream>>>(BFY, 1024, fc2T, 1024,
      NTOK, 256, 1024, fc2_b, nullptr, T1, nullptr, 256);               // fc2 + residual
  cast_f32_bf16_kernel<<<96000, 256, 0, stream>>>(T1, BFX);             // tokens -> bf16
  gemm_bf16_wmma<4, 128><<<750, 256, 0, stream>>>(BFX, 256, weff, 256,
      NTOK, 64, 256, cst, nullptr, (float*)d_out, nullptr, 0);          // folded tail -> NCHW out
}
